// sLSTMBlock_16561393893827
// MI455X (gfx1250) — matrix operand order repaired and checked by reading, not compiled, not run on hardware
//
#include <hip/hip_runtime.h>
#include <hip/hip_bf16.h>
#include <stdint.h>

// ---------------------------------------------------------------------------
// sLSTM cell step for MI455X (gfx1250), bf16 WMMA matrix-core path.
//   A = [x | h_prev]  : 4096 x 4096, packed into WMMA A-fragment tile order
//   B[g] = [Wg | Rg]  : 2048(N) x 4096(K), packed into WMMA B-fragment order
// Packed tile = 32 lanes x 32 bytes (16 bf16) in contiguous K order for A/B
// (lane half h: K = 16h+0..15), so GEMM fragment loads are fully coalesced streaming b128s.
// GEMM: block 256 thr = 8 waves (2M x 4N), block tile 128x128, wave tile
// 64M x 32N for each of the 4 gates (32 WMMA per 24 b128 loads per K-step).
// Epilogue (tanh / sigmoid / stabilized exp gating) fused in registers.
// ---------------------------------------------------------------------------

typedef __attribute__((ext_vector_type(16))) __bf16 v16bf;
typedef __attribute__((ext_vector_type(8)))  float  v8f;

#define BATCH 4096
#define INSZ  2048
#define HID   2048
#define KDIM  4096            // INSZ + HID
#define KTILES (KDIM / 32)    // 128
#define TILE_ELEMS 512        // 32 lanes * 16 bf16 per packed 16x32 tile

__device__ __forceinline__ unsigned short f32_to_bf16_rne(float f) {
    union { float f; uint32_t u; } v; v.f = f;
    uint32_t u = v.u;
    uint32_t r = u + 0x7FFFu + ((u >> 16) & 1u);
    return (unsigned short)(r >> 16);
}

union PackOut { unsigned short s[16]; uint4 u[2]; };

// Pack [x | h_prev] -> A fragments. One thread = one lane's 32B chunk.
// Tile index ti = mt*KTILES + kt ; A-layout (16-bit, 16x32):
//   lane half 0: K = k0+0..15 ; half 1: K = k0+16..31
__global__ __launch_bounds__(256)
void packA_kernel(const float* __restrict__ x, const float* __restrict__ h,
                  unsigned short* __restrict__ A) {
    const uint32_t t    = blockIdx.x * 256u + threadIdx.x;
    const uint32_t ti   = t >> 5;
    const uint32_t lane = t & 31u;
    const uint32_t mt   = ti >> 7;          // / KTILES
    const uint32_t kt   = ti & 127u;
    const uint32_t row  = (mt << 4) + (lane & 15u);
    const uint32_t half = lane >> 4;
    const uint32_t k0   = kt * 32u;
    // whole 32-wide K block lives on one side of the x|h split (2048 % 32 == 0)
    const float* src = (k0 < INSZ) ? (x + (size_t)row * INSZ + k0)
                                   : (h + (size_t)row * HID + (k0 - INSZ));
    const float4 g0a = *reinterpret_cast<const float4*>(src + half * 16);
    const float4 g0b = *reinterpret_cast<const float4*>(src + half * 16 + 4);
    const float4 g1a = *reinterpret_cast<const float4*>(src + half * 16 + 8);
    const float4 g1b = *reinterpret_cast<const float4*>(src + half * 16 + 12);
    PackOut o;
    o.s[0] = f32_to_bf16_rne(g0a.x); o.s[1] = f32_to_bf16_rne(g0a.y);
    o.s[2] = f32_to_bf16_rne(g0a.z); o.s[3] = f32_to_bf16_rne(g0a.w);
    o.s[4] = f32_to_bf16_rne(g0b.x); o.s[5] = f32_to_bf16_rne(g0b.y);
    o.s[6] = f32_to_bf16_rne(g0b.z); o.s[7] = f32_to_bf16_rne(g0b.w);
    o.s[8]  = f32_to_bf16_rne(g1a.x); o.s[9]  = f32_to_bf16_rne(g1a.y);
    o.s[10] = f32_to_bf16_rne(g1a.z); o.s[11] = f32_to_bf16_rne(g1a.w);
    o.s[12] = f32_to_bf16_rne(g1b.x); o.s[13] = f32_to_bf16_rne(g1b.y);
    o.s[14] = f32_to_bf16_rne(g1b.z); o.s[15] = f32_to_bf16_rne(g1b.w);
    uint4* dst = reinterpret_cast<uint4*>(A + (size_t)ti * TILE_ELEMS + lane * 16u);
    dst[0] = o.u[0];
    dst[1] = o.u[1];
}

// Pack [Wg | Rg] -> B fragments. Tile index ti = ((g*128 + nt)*KTILES + kt).
// B-layout (16-bit, 32x16): lane half 0: K = k0+0..15 ; half 1: K = k0+16..31.
__global__ __launch_bounds__(256)
void packB_kernel(const float* __restrict__ Wz, const float* __restrict__ Wi,
                  const float* __restrict__ Wf, const float* __restrict__ Wo,
                  const float* __restrict__ Rz, const float* __restrict__ Ri,
                  const float* __restrict__ Rf, const float* __restrict__ Ro,
                  unsigned short* __restrict__ B) {
    const uint32_t t    = blockIdx.x * 256u + threadIdx.x;
    const uint32_t ti   = t >> 5;
    const uint32_t lane = t & 31u;
    const uint32_t g    = ti >> 14;            // / (128 * KTILES)
    const uint32_t rem  = ti & 16383u;
    const uint32_t nt   = rem >> 7;
    const uint32_t kt   = rem & 127u;
    const uint32_t ncol = (nt << 4) + (lane & 15u);
    const uint32_t half = lane >> 4;
    const uint32_t k0   = kt * 32u + half * 16u;
    const float* W = (g == 0) ? Wz : (g == 1) ? Wi : (g == 2) ? Wf : Wo;
    const float* R = (g == 0) ? Rz : (g == 1) ? Ri : (g == 2) ? Rf : Ro;
    const float* src = (k0 < INSZ) ? (W + (size_t)ncol * INSZ + k0)
                                   : (R + (size_t)ncol * HID + (k0 - INSZ));
    PackOut o;
#pragma unroll
    for (int q = 0; q < 4; ++q) {
        const float4 v = *reinterpret_cast<const float4*>(src + q * 4);
        o.s[q * 4 + 0] = f32_to_bf16_rne(v.x);
        o.s[q * 4 + 1] = f32_to_bf16_rne(v.y);
        o.s[q * 4 + 2] = f32_to_bf16_rne(v.z);
        o.s[q * 4 + 3] = f32_to_bf16_rne(v.w);
    }
    uint4* dst = reinterpret_cast<uint4*>(B + (size_t)ti * TILE_ELEMS + lane * 16u);
    dst[0] = o.u[0];
    dst[1] = o.u[1];
}

union Frag { v16bf v; uint4 u[2]; };

// Fused 4-gate GEMM + sLSTM epilogue.
// 8 waves = 2(M) x 4(N); wave tile 64M x 32N per gate -> acc[4][4][2] v8f.
__global__ __launch_bounds__(256)
void slstm_gemm_kernel(const unsigned short* __restrict__ A,
                       const unsigned short* __restrict__ B,
                       const float* __restrict__ bWz, const float* __restrict__ bWi,
                       const float* __restrict__ bWf, const float* __restrict__ bWo,
                       const float* __restrict__ bRz, const float* __restrict__ bRi,
                       const float* __restrict__ bRf, const float* __restrict__ bRo,
                       const float* __restrict__ c_prev, const float* __restrict__ n_prev,
                       const float* __restrict__ m_prev,
                       float* __restrict__ outH, float* __restrict__ outC,
                       float* __restrict__ outN, float* __restrict__ outM) {
    const int lane  = threadIdx.x & 31;
    const int wave  = threadIdx.x >> 5;
    const int waveM = wave & 1;    // 0..1
    const int waveN = wave >> 1;   // 0..3
    const int l16   = lane & 15;
    const int half  = lane >> 4;

    const int m0 = blockIdx.y * 128 + waveM * 64;
    const int n0 = blockIdx.x * 128 + waveN * 32;
    const int mtBase = m0 >> 4;    // 4 consecutive M tiles
    const int ntBase = n0 >> 4;    // 2 consecutive N tiles

    // Per-lane packed-fragment base pointers; K-step stride = TILE_ELEMS.
    const unsigned short* aP[4];
#pragma unroll
    for (int mi = 0; mi < 4; ++mi)
        aP[mi] = A + ((size_t)(mtBase + mi) * KTILES) * TILE_ELEMS + (size_t)lane * 16;

    const unsigned short* bP[4][2];
#pragma unroll
    for (int g = 0; g < 4; ++g)
#pragma unroll
        for (int ni = 0; ni < 2; ++ni)
            bP[g][ni] = B + (((size_t)g * 128 + (size_t)(ntBase + ni)) * KTILES) * TILE_ELEMS
                          + (size_t)lane * 16;

    v8f acc[4][4][2];
#pragma unroll
    for (int g = 0; g < 4; ++g)
#pragma unroll
        for (int mi = 0; mi < 4; ++mi)
#pragma unroll
            for (int ni = 0; ni < 2; ++ni)
                acc[g][mi][ni] = (v8f){};

    for (int kt = 0; kt < KTILES; ++kt) {
        const size_t koff = (size_t)kt * TILE_ELEMS;
        Frag af[4];
#pragma unroll
        for (int mi = 0; mi < 4; ++mi) {
            const uint4* p = reinterpret_cast<const uint4*>(aP[mi] + koff);
            af[mi].u[0] = p[0];
            af[mi].u[1] = p[1];
        }
        Frag bfr[4][2];
#pragma unroll
        for (int g = 0; g < 4; ++g)
#pragma unroll
            for (int ni = 0; ni < 2; ++ni) {
                const uint4* p = reinterpret_cast<const uint4*>(bP[g][ni] + koff);
                bfr[g][ni].u[0] = p[0];
                bfr[g][ni].u[1] = p[1];
            }
#pragma unroll
        for (int g = 0; g < 4; ++g)
#pragma unroll
            for (int mi = 0; mi < 4; ++mi)
#pragma unroll
                for (int ni = 0; ni < 2; ++ni)
                    acc[g][mi][ni] = __builtin_amdgcn_wmma_f32_16x16x32_bf16(
                        false, af[mi].v, false, bfr[g][ni].v,
                        (short)0, acc[g][mi][ni], false, false);
    }

    // ---------------- fused sLSTM epilogue ----------------
    // C/D layout: VGPR r, lanes 0-15 -> (M=r, N=l16); lanes 16-31 -> (M=8+r, N=l16).
#pragma unroll
    for (int ni = 0; ni < 2; ++ni) {
        const int col = n0 + ni * 16 + l16;
        const float bz = bWz[col] + bRz[col];
        const float bi = bWi[col] + bRi[col];
        const float bf = bWf[col] + bRf[col];
        const float bo = bWo[col] + bRo[col];
#pragma unroll
        for (int mi = 0; mi < 4; ++mi) {
#pragma unroll
            for (int r = 0; r < 8; ++r) {
                const int row = m0 + mi * 16 + half * 8 + r;
                const size_t idx = (size_t)row * HID + col;
                const float zp = acc[0][mi][ni][r] + bz;
                const float ip = acc[1][mi][ni][r] + bi;
                const float fp = acc[2][mi][ni][r] + bf;
                const float op = acc[3][mi][ni][r] + bo;

                const float mp = m_prev[idx];
                const float cp = c_prev[idx];
                const float np = n_prev[idx];

                const float z    = tanhf(zp);
                const float o    = 1.0f / (1.0f + __expf(-op));
                const float mnew = fmaxf(fp + mp, ip);
                const float ih   = __expf(ip - mnew);
                const float fh   = __expf(fp + mp - mnew);
                const float c    = fh * cp + ih * z;
                const float n    = fh * np + ih;
                const float hh   = o * c / n;

                outH[idx] = hh;
                outC[idx] = c;
                outN[idx] = n;
                outM[idx] = mnew;
            }
        }
    }
}

extern "C" void kernel_launch(void* const* d_in, const int* in_sizes, int n_in,
                              void* d_out, int out_size, void* d_ws, size_t ws_size,
                              hipStream_t stream) {
    const float* x      = (const float*)d_in[0];
    const float* h_prev = (const float*)d_in[1];
    const float* c_prev = (const float*)d_in[2];
    const float* n_prev = (const float*)d_in[3];
    const float* m_prev = (const float*)d_in[4];
    const float* Wz_w = (const float*)d_in[5];  const float* Wz_b = (const float*)d_in[6];
    const float* Wi_w = (const float*)d_in[7];  const float* Wi_b = (const float*)d_in[8];
    const float* Wf_w = (const float*)d_in[9];  const float* Wf_b = (const float*)d_in[10];
    const float* Wo_w = (const float*)d_in[11]; const float* Wo_b = (const float*)d_in[12];
    const float* Rz_w = (const float*)d_in[13]; const float* Rz_b = (const float*)d_in[14];
    const float* Ri_w = (const float*)d_in[15]; const float* Ri_b = (const float*)d_in[16];
    const float* Rf_w = (const float*)d_in[17]; const float* Rf_b = (const float*)d_in[18];
    const float* Ro_w = (const float*)d_in[19]; const float* Ro_b = (const float*)d_in[20];

    // Workspace: packed A (32 MB) + packed B (64 MB) = 96 MB of bf16 operands.
    unsigned short* A = (unsigned short*)d_ws;
    unsigned short* B = A + (size_t)BATCH * KDIM;

    {   // pack A = [x | h_prev] into fragment tiles: 256*128 tiles * 32 lanes
        const int threads = (BATCH / 16) * KTILES * 32;     // 1,048,576
        packA_kernel<<<threads / 256, 256, 0, stream>>>(x, h_prev, A);
    }
    {   // pack B[g] = [Wg | Rg]: 4*128*128 tiles * 32 lanes
        const int threads = 4 * (HID / 16) * KTILES * 32;   // 2,097,152
        packB_kernel<<<threads / 256, 256, 0, stream>>>(Wz_w, Wi_w, Wf_w, Wo_w,
                                                        Rz_w, Ri_w, Rf_w, Ro_w, B);
    }

    float* out = (float*)d_out;
    const size_t NEL = (size_t)BATCH * HID;                 // 8388608
    dim3 grid(HID / 128, BATCH / 128);                      // (16, 32)
    slstm_gemm_kernel<<<grid, 256, 0, stream>>>(
        A, B,
        Wz_b, Wi_b, Wf_b, Wo_b,
        Rz_b, Ri_b, Rf_b, Ro_b,
        c_prev, n_prev, m_prev,
        out, out + NEL, out + 2 * NEL, out + 3 * NEL);
}
